// Multihead_78597901516986
// MI455X (gfx1250) — matrix operand order repaired and checked by reading, not compiled, not run on hardware
//
#include <hip/hip_runtime.h>
#include <hip/hip_bf16.h>

// ---------------------------------------------------------------------------
// Problem constants (B, S, E, H) = (2, 2048, 1024, 16), D = 64
// ---------------------------------------------------------------------------
#define BB 2
#define SS 2048
#define EE 1024
#define HH 16
#define DD 64
#define N3E (3 * EE)          // 3072
#define MROWS (BB * SS)       // 4096
#define QKV_STRIDE ((size_t)BB * HH * SS * DD)  // 4,194,304 halves per tensor

typedef __attribute__((ext_vector_type(16))) _Float16 v16h;
typedef __attribute__((ext_vector_type(8)))  _Float16 v8h;
typedef __attribute__((ext_vector_type(8)))  float    v8f;

__device__ __forceinline__ v8f zero_v8f() {
    v8f z = {0.f, 0.f, 0.f, 0.f, 0.f, 0.f, 0.f, 0.f};
    return z;
}

// A-fragment (16x32 f16), row-major LDS tile with row stride `ld` halves
// (ld*2 bytes must be a multiple of 16 for the b128 path).
// Lane L: M = L&15, k-base = (L>=16)?8:0; halves 0..7 -> kb+j, 8..15 -> kb+16+j.
__device__ __forceinline__ v16h ldsA(const _Float16* base, int ld, int k0, int lane) {
    const int m  = lane & 15;
    const int kb = (lane >> 4) * 8;
    const _Float16* p = base + m * ld + k0 + kb;
    const v8h lo = *(const v8h*)(p);
    const v8h hi = *(const v8h*)(p + 16);
    return __builtin_shufflevector(lo, hi, 0, 1, 2, 3, 4, 5, 6, 7,
                                           8, 9, 10, 11, 12, 13, 14, 15);
}

// B-fragment (32x16 f16) from a COLUMN-MAJOR LDS tile: Bs[n][k], row stride
// `ld` halves.  Lane L: N = L&15, k-base = (L>=16)?8:0; halves 0..7 -> kb+j, 8..15 -> kb+16+j.
// Two 16-byte chunks per lane -> two ds_load_b128.
__device__ __forceinline__ v16h ldsBcm(const _Float16* base, int ld, int k0, int lane) {
    const int n  = lane & 15;
    const int kb = (lane >> 4) * 8;
    const _Float16* p = base + n * ld + k0 + kb;
    const v8h lo = *(const v8h*)(p);
    const v8h hi = *(const v8h*)(p + 16);
    return __builtin_shufflevector(lo, hi, 0, 1, 2, 3, 4, 5, 6, 7,
                                           8, 9, 10, 11, 12, 13, 14, 15);
}

// ---------------------------------------------------------------------------
// Kernel 1: fused QKV projection.  qkv[m][n] = sum_e x[m][e]*w[n][e] + b[n]
// M=4096, N=3072, K=1024.  Block: 256 thr (8 waves), tile 64x128, BK=32.
// Q,K stored [b][h][s][d]; V stored TRANSPOSED [b][h][d][s] (for kernel 4).
// ---------------------------------------------------------------------------
__global__ __launch_bounds__(256) void qkv_gemm_kernel(
    const float* __restrict__ x, const float* __restrict__ w,
    const float* __restrict__ bias, _Float16* __restrict__ qkv) {
    __shared__ _Float16 As[64][32];        // row-major, ld=32 (64 B rows)
    __shared__ _Float16 Bs[128][40];       // column-major [n][k], ld=40 (80 B rows)

    const int m0   = blockIdx.x * 64;      // gridDim.x = 64
    const int n0   = blockIdx.y * 128;     // gridDim.y = 24
    const int t    = threadIdx.x;
    const int lane = t & 31;
    const int wid  = t >> 5;
    const int wm   = (wid & 3) * 16;       // wave M offset
    const int wn   = (wid >> 2) * 64;      // wave N offset

    v8f acc[4];
#pragma unroll
    for (int i = 0; i < 4; ++i) acc[i] = zero_v8f();

    for (int k0 = 0; k0 < EE; k0 += 32) {
        {   // stage A (64x32), f32 -> f16; contiguous load + contiguous store
            const int r = t >> 2, c = (t & 3) * 8;
            const float* src = x + (size_t)(m0 + r) * EE + k0 + c;
#pragma unroll
            for (int j = 0; j < 8; ++j) As[r][c + j] = (_Float16)src[j];
        }
        {   // stage B column-major: Bs[n][kk..kk+15] = w[n0+n][k0+kk..]
            const int n = t >> 1, kk = (t & 1) * 16;
            const float* src = w + (size_t)(n0 + n) * EE + k0 + kk;
#pragma unroll
            for (int j = 0; j < 16; ++j) Bs[n][kk + j] = (_Float16)src[j];
        }
        __syncthreads();

        // Preload ALL fragments, then issue the WMMA chain back-to-back so the
        // ds_load_b128s pipeline under a single dscnt wait.
        const v16h a = ldsA(&As[wm][0], 32, 0, lane);
        v16h bf[4];
#pragma unroll
        for (int i = 0; i < 4; ++i)
            bf[i] = ldsBcm(&Bs[wn + i * 16][0], 40, 0, lane);
#pragma unroll
        for (int i = 0; i < 4; ++i)
            acc[i] = __builtin_amdgcn_wmma_f32_16x16x32_f16(
                false, a, false, bf[i], (short)0, acc[i], false, false);
        __syncthreads();
    }

    // bias + scatter: Q,K -> [b][h][s][d] f16, V -> [b][h][d][s] f16
    // (branchless index select to avoid exec-mask divergence)
    const int ln = lane & 15, lm = (lane >> 4) * 8;
#pragma unroll
    for (int i = 0; i < 4; ++i) {
        const int n     = n0 + wn + i * 16 + ln;
        const int which = n >> 10;            // 0..2
        const int h     = (n & 1023) >> 6;    // 0..15
        const int d     = n & 63;
        const float bv  = bias[n];
#pragma unroll
        for (int r = 0; r < 8; ++r) {
            const int m = m0 + wm + lm + r;
            const int b = m >> 11;            // / 2048
            const int s = m & 2047;
            const size_t bhh  = (size_t)(b * HH + h);
            const size_t idxN = (size_t)which * QKV_STRIDE + ((bhh * SS + s) * DD + d);
            const size_t idxV = 2 * QKV_STRIDE + ((bhh * DD + d) * SS + s);
            qkv[which == 2 ? idxV : idxN] = (_Float16)(acc[i][r] + bv);
        }
    }
}

// ---------------------------------------------------------------------------
// Kernel 2: raw attention logits.  scores = Q @ K^T per (b,h).
// M=N=2048, K=64.  Block tile 64x128; whole K staged (2 WMMA K-steps).
// K tile staged column-major Ks[n][d] (contiguous both sides since K=[s][d]).
// ---------------------------------------------------------------------------
__global__ __launch_bounds__(256) void scores_kernel(
    const _Float16* __restrict__ qkv, float* __restrict__ attn) {
    __shared__ _Float16 Qs[64][72];        // row-major (144 B rows)
    __shared__ _Float16 Ks[128][72];       // column-major [n][d] (144 B rows)

    const int m0   = blockIdx.x * 64;      // 32 tiles
    const int n0   = blockIdx.y * 128;     // 16 tiles
    const int bh   = blockIdx.z;           // 32
    const int t    = threadIdx.x;
    const int lane = t & 31;
    const int wid  = t >> 5;
    const int wm   = (wid & 3) * 16;
    const int wn   = (wid >> 2) * 64;

    const _Float16* q = qkv + (size_t)bh * SS * DD;
    const _Float16* k = qkv + QKV_STRIDE + (size_t)bh * SS * DD;

    {   // stage Q tile 64x64 (contiguous)
        const int r = t >> 2, c = (t & 3) * 16;
        const _Float16* src = q + (size_t)(m0 + r) * DD + c;
#pragma unroll
        for (int j = 0; j < 16; ++j) Qs[r][c + j] = src[j];
    }
    {   // stage K tile column-major: Ks[n][d] = K[n0+n][d] (contiguous)
        const int n = t >> 1, d0 = (t & 1) * 32;
        const _Float16* src = k + (size_t)(n0 + n) * DD + d0;
#pragma unroll
        for (int j = 0; j < 32; ++j) Ks[n][d0 + j] = src[j];
    }
    __syncthreads();

    v8f acc[4];
#pragma unroll
    for (int i = 0; i < 4; ++i) acc[i] = zero_v8f();

#pragma unroll
    for (int ks = 0; ks < 64; ks += 32) {
        const v16h a = ldsA(&Qs[wm][0], 72, ks, lane);
        v16h bf[4];
#pragma unroll
        for (int i = 0; i < 4; ++i)
            bf[i] = ldsBcm(&Ks[wn + i * 16][0], 72, ks, lane);
#pragma unroll
        for (int i = 0; i < 4; ++i)
            acc[i] = __builtin_amdgcn_wmma_f32_16x16x32_f16(
                false, a, false, bf[i], (short)0, acc[i], false, false);
    }

    const int ln = lane & 15, lm = (lane >> 4) * 8;
#pragma unroll
    for (int i = 0; i < 4; ++i) {
        const int col = n0 + wn + i * 16 + ln;
#pragma unroll
        for (int r = 0; r < 8; ++r) {
            const int row = m0 + wm + lm + r;
            attn[((size_t)bh * SS + row) * SS + col] = acc[i][r];
        }
    }
}

// ---------------------------------------------------------------------------
// Kernel 3: in-place row softmax over attn (rows of length 2048).
// ---------------------------------------------------------------------------
__device__ __forceinline__ float wave_rmax(float v) {
#pragma unroll
    for (int off = 16; off > 0; off >>= 1) v = fmaxf(v, __shfl_xor(v, off, 32));
    return v;
}
__device__ __forceinline__ float wave_rsum(float v) {
#pragma unroll
    for (int off = 16; off > 0; off >>= 1) v += __shfl_xor(v, off, 32);
    return v;
}

__global__ __launch_bounds__(256) void softmax_kernel(float* __restrict__ attn) {
    __shared__ float red[8];
    const size_t row = blockIdx.x;                 // B*H*S rows
    float* p = attn + row * (size_t)SS;
    const int t = threadIdx.x, lane = t & 31, wid = t >> 5;

    float vals[8];
    float mx = -1e30f;
#pragma unroll
    for (int j = 0; j < 8; ++j) {
        vals[j] = p[t + j * 256];
        mx = fmaxf(mx, vals[j]);
    }
    mx = wave_rmax(mx);
    if (lane == 0) red[wid] = mx;
    __syncthreads();
    float bm = red[0];
#pragma unroll
    for (int i = 1; i < 8; ++i) bm = fmaxf(bm, red[i]);
    __syncthreads();

    float sum = 0.f;
#pragma unroll
    for (int j = 0; j < 8; ++j) {
        vals[j] = __expf(vals[j] - bm);
        sum += vals[j];
    }
    sum = wave_rsum(sum);
    if (lane == 0) red[wid] = sum;
    __syncthreads();
    float bs = 0.f;
#pragma unroll
    for (int i = 0; i < 8; ++i) bs += red[i];
    const float inv = __frcp_rn(bs);
#pragma unroll
    for (int j = 0; j < 8; ++j) p[t + j * 256] = vals[j] * inv;
}

// ---------------------------------------------------------------------------
// Kernel 4: values = attn @ V per (b,h).  M=2048, N=64, K=2048.
// attn read f32 -> f16 fragments; V already transposed [b][h][d][s] in ws, so
// the column-major Vs[d][k] staging is contiguous on both sides.
// ---------------------------------------------------------------------------
__global__ __launch_bounds__(256) void attn_v_kernel(
    const float* __restrict__ attn, const _Float16* __restrict__ qkv,
    float* __restrict__ values) {
    __shared__ _Float16 Ap[64][40];        // row-major attn tile (80 B rows)
    __shared__ _Float16 Vs[64][40];        // column-major [d][k] (80 B rows)

    const int m0   = blockIdx.x * 64;      // 32 tiles
    const int bh   = blockIdx.y;           // 32
    const int b    = bh >> 4;
    const int h    = bh & 15;
    const int t    = threadIdx.x;
    const int lane = t & 31;
    const int wid  = t >> 5;
    const int wm   = (wid & 3) * 16;
    const int wn   = (wid >> 2) * 32;      // two waves split N=64

    const float*    arow = attn + (size_t)bh * SS * SS;
    const _Float16* vt   = qkv + 2 * QKV_STRIDE + (size_t)bh * DD * SS;

    v8f acc[2];
#pragma unroll
    for (int i = 0; i < 2; ++i) acc[i] = zero_v8f();

    for (int k0 = 0; k0 < SS; k0 += 32) {
        {   // stage attn tile 64x32, f32 -> f16 (contiguous)
            const int r = t >> 2, c = (t & 3) * 8;
            const float* src = arow + (size_t)(m0 + r) * SS + k0 + c;
#pragma unroll
            for (int j = 0; j < 8; ++j) Ap[r][c + j] = (_Float16)src[j];
        }
        {   // stage V^T tile: Vs[d][kk..kk+7] = Vt[d][k0+kk..] (contiguous)
            const int d = t >> 2, c = (t & 3) * 8;
            const _Float16* src = vt + (size_t)d * SS + k0 + c;
#pragma unroll
            for (int j = 0; j < 8; ++j) Vs[d][c + j] = src[j];
        }
        __syncthreads();

        const v16h a = ldsA(&Ap[wm][0], 40, 0, lane);
        v16h bf[2];
#pragma unroll
        for (int i = 0; i < 2; ++i)
            bf[i] = ldsBcm(&Vs[wn + i * 16][0], 40, 0, lane);
#pragma unroll
        for (int i = 0; i < 2; ++i)
            acc[i] = __builtin_amdgcn_wmma_f32_16x16x32_f16(
                false, a, false, bf[i], (short)0, acc[i], false, false);
        __syncthreads();
    }

    const int ln = lane & 15, lm = (lane >> 4) * 8;
#pragma unroll
    for (int i = 0; i < 2; ++i) {
        const int d = wn + i * 16 + ln;
#pragma unroll
        for (int r = 0; r < 8; ++r) {
            const int s = m0 + wm + lm + r;
            values[((size_t)(b * SS + s)) * EE + h * DD + d] = acc[i][r];
        }
    }
}

// ---------------------------------------------------------------------------
// Launch
// ---------------------------------------------------------------------------
extern "C" void kernel_launch(void* const* d_in, const int* in_sizes, int n_in,
                              void* d_out, int out_size, void* d_ws, size_t ws_size,
                              hipStream_t stream) {
    const float* x    = (const float*)d_in[0];   // (B, S, E)
    const float* w    = (const float*)d_in[1];   // (3E, E)
    const float* bias = (const float*)d_in[2];   // (3E,)
    // d_in[3] = heads (scalar), fixed at 16 here.

    float* attn   = (float*)d_out;                         // (B,H,1,S,S)
    float* values = attn + (size_t)BB * HH * SS * SS;      // (B,S,E)

    _Float16* qkv = (_Float16*)d_ws;  // needs 3 * 4,194,304 * 2 B = 24 MB

    qkv_gemm_kernel<<<dim3(MROWS / 64, N3E / 128), 256, 0, stream>>>(x, w, bias, qkv);
    scores_kernel<<<dim3(SS / 64, SS / 128, BB * HH), 256, 0, stream>>>(qkv, attn);
    softmax_kernel<<<dim3(BB * HH * SS), 256, 0, stream>>>(attn);
    attn_v_kernel<<<dim3(SS / 64, BB * HH), 256, 0, stream>>>(attn, qkv, values);
}
